// Model_25744033972910
// MI455X (gfx1250) — matrix operand order repaired and checked by reading, not compiled, not run on hardware
//
#include <hip/hip_runtime.h>
#include <stdint.h>

#define BB 8
#define CC 256
#define NN 2048
#define MM 4096
#define KNN 10

typedef __attribute__((ext_vector_type(16))) __bf16 v16bf;
typedef __attribute__((ext_vector_type(8)))  float  v8f;

// ---------- helpers ----------
__device__ __forceinline__ unsigned short f2bf(float f) {
    unsigned int u = __float_as_uint(f);
    unsigned int r = u + 0x7FFFu + ((u >> 16) & 1u);   // round-to-nearest-even
    return (unsigned short)(r >> 16);
}
__device__ __forceinline__ float bf2f(unsigned short b) {
    return __uint_as_float(((unsigned int)b) << 16);
}

union FragU { uint4 q[2]; v16bf v; };

// A fragment: 16(M)x32(K) bf16, row-major A (lda = K elems). Lane l=M row,
// half h: K in {h*8..h*8+7} and {16+h*8..16+h*8+7}  (ISA 7.12.2 16-bit A layout)
__device__ __forceinline__ v16bf ldA(const __bf16* arow, int k0, int h) {
    FragU u;
    u.q[0] = *(const uint4*)(arow + k0 + h * 8);
    u.q[1] = *(const uint4*)(arow + k0 + 16 + h * 8);
    return u.v;
}
// B fragment: 32(K)x16(N) bf16 from N-major (row n holds K contiguous).
// Lane l = N col, half h holds K in {h*8..h*8+7} and {16+h*8..16+h*8+7}.
__device__ __forceinline__ v16bf ldB(const __bf16* brow, int k0, int h) {
    FragU u;
    const __bf16* p = brow + k0 + h * 8;
    u.q[0] = *(const uint4*)(p);
    u.q[1] = *(const uint4*)(p + 16);
    return u.v;
}

// 16(M) x 64(N) strip, both operands K-contiguous rows ("TN").
template <int KTOT>
__device__ __forceinline__ void gemm_strip16x64(const __bf16* A, int lda,
                                                const __bf16* Bm, int ldb,
                                                int m0, int n0, int l, int h,
                                                v8f acc[4]) {
    const __bf16* arow = A + (size_t)(m0 + l) * lda;
    for (int k0 = 0; k0 < KTOT; k0 += 32) {
        v16bf av = ldA(arow, k0, h);
#pragma unroll
        for (int j = 0; j < 4; ++j) {
            const __bf16* brow = Bm + (size_t)(n0 + j * 16 + l) * ldb;
            v16bf bv = ldB(brow, k0, h);
            acc[j] = __builtin_amdgcn_wmma_f32_16x16x32_bf16(
                false, av, false, bv, (short)0, acc[j], false, false);
        }
    }
}

// ---------- K1: transpose x (B,C,N) f32 -> xT (B,N,C) f32 + bf16 ----------
__global__ void __launch_bounds__(256)
k_transpose_x(const float* __restrict__ x, float* __restrict__ xT,
              unsigned short* __restrict__ xTb) {
    __shared__ float tile[32][33];
    int b = blockIdx.z;
    int n0 = blockIdx.x * 32, c0 = blockIdx.y * 32;
    int tx = threadIdx.x, ty = threadIdx.y;  // 32 x 8
    const float* xb = x + (size_t)b * CC * NN;
#pragma unroll
    for (int i = 0; i < 4; ++i)
        tile[ty + i * 8][tx] = xb[(size_t)(c0 + ty + i * 8) * NN + n0 + tx];
    __syncthreads();
    float* xr = xT + (size_t)b * NN * CC;
    unsigned short* xbr = xTb + (size_t)b * NN * CC;
#pragma unroll
    for (int i = 0; i < 4; ++i) {
        float v = tile[tx][ty + i * 8];
        size_t o = (size_t)(n0 + ty + i * 8) * CC + c0 + tx;
        xr[o] = v;
        xbr[o] = f2bf(v);
    }
}

// ---------- K2: convert three CxC weights to bf16 ----------
__global__ void __launch_bounds__(256)
k_convert_w(const float* a, const float* b, const float* c,
            unsigned short* oa, unsigned short* ob, unsigned short* oc) {
    int i = blockIdx.x * 256 + threadIdx.x;  // 0..65535
    oa[i] = f2bf(a[i]);
    ob[i] = f2bf(b[i]);
    oc[i] = f2bf(c[i]);
}

// ---------- K3: generic K=256 bf16 GEMM, out bf16 row-major [M][N] ----------
__global__ void __launch_bounds__(256)
k_gemm_k256(const unsigned short* A_, size_t strideA, int lda,
            const unsigned short* B_, size_t strideB, int ldb,
            const float* bias,  // per-M row, may be null
            unsigned short* out, size_t strideO, int ldo, int nstrips) {
    int b = blockIdx.y;
    int wave = (blockIdx.x * blockDim.x + threadIdx.x) >> 5;
    int lane = threadIdx.x & 31, l = lane & 15, h = lane >> 4;
    int m0 = (wave / nstrips) * 16;
    int n0 = (wave % nstrips) * 64;
    const __bf16* A = (const __bf16*)A_ + strideA * b;
    const __bf16* Bm = (const __bf16*)B_ + strideB * b;
    v8f acc[4] = {};
    gemm_strip16x64<CC>(A, lda, Bm, ldb, m0, n0, l, h, acc);
    unsigned short* ob = out + strideO * b;
#pragma unroll
    for (int j = 0; j < 4; ++j)
#pragma unroll
        for (int r = 0; r < 8; ++r) {
            int m = m0 + r + h * 8;
            float v = acc[j][r] + (bias ? bias[m] : 0.0f);
            ob[(size_t)m * ldo + n0 + j * 16 + l] = f2bf(v);
        }
}

// ---------- K4: softmax row stats (online, one energy sweep) ----------
// Lane owns a fixed column n of energy tiles D[m][n]; online max/sum over m.
__global__ void __launch_bounds__(256)
k_row_stats(const unsigned short* yT_, float* rowmax, float* rowsum) {
    int b = blockIdx.y;
    int wave = (blockIdx.x * blockDim.x + threadIdx.x) >> 5;
    int lane = threadIdx.x & 31, l = lane & 15, h = lane >> 4;
    int n0 = wave * 16;
    const __bf16* Y = (const __bf16*)yT_ + (size_t)b * NN * CC;
    v16bf bfr[8];
    const __bf16* brow = Y + (size_t)(n0 + l) * CC;
#pragma unroll
    for (int kk = 0; kk < 8; ++kk) bfr[kk] = ldB(brow, kk * 32, h);
    float rmax = -3.0e38f, rsum = 0.0f;
    for (int mt = 0; mt < NN / 16; ++mt) {
        const __bf16* arow = Y + (size_t)(mt * 16 + l) * CC;
        v8f acc = {};
#pragma unroll
        for (int kk = 0; kk < 8; ++kk) {
            v16bf av = ldA(arow, kk * 32, h);
            acc = __builtin_amdgcn_wmma_f32_16x16x32_bf16(
                false, av, false, bfr[kk], (short)0, acc, false, false);
        }
        float tmax = acc[0];
#pragma unroll
        for (int r = 1; r < 8; ++r) tmax = fmaxf(tmax, acc[r]);
        float nm = fmaxf(rmax, tmax);
        rsum *= __expf(rmax - nm);
        rmax = nm;
#pragma unroll
        for (int r = 0; r < 8; ++r) rsum += __expf(acc[r] - rmax);
    }
    float om = __shfl_xor(rmax, 16, 32);
    float os = __shfl_xor(rsum, 16, 32);
    float fm = fmaxf(rmax, om);
    float fs = rsum * __expf(rmax - fm) + os * __expf(om - fm);
    if (lane < 16) {
        rowmax[(size_t)b * NN + n0 + l] = fm;
        rowsum[(size_t)b * NN + n0 + l] = fs;
    }
}

// ---------- K5: attnT[m][n] = exp(e[m,n]-rowmax[n])/rowsum[n], bf16 ----------
__global__ void __launch_bounds__(256)
k_attn(const unsigned short* yT_, const float* rowmax, const float* rowsum,
       unsigned short* attnT) {
    int b = blockIdx.y;
    int wave = (blockIdx.x * blockDim.x + threadIdx.x) >> 5;
    int lane = threadIdx.x & 31, l = lane & 15, h = lane >> 4;
    int m0 = (wave / 32) * 16;
    int n0 = (wave % 32) * 64;
    const __bf16* Y = (const __bf16*)yT_ + (size_t)b * NN * CC;
    v8f acc[4] = {};
    gemm_strip16x64<CC>(Y, CC, Y, CC, m0, n0, l, h, acc);
    unsigned short* At = attnT + (size_t)b * NN * NN;
#pragma unroll
    for (int j = 0; j < 4; ++j) {
        int n = n0 + j * 16 + l;
        float rm = rowmax[(size_t)b * NN + n];
        float ri = 1.0f / rowsum[(size_t)b * NN + n];
#pragma unroll
        for (int r = 0; r < 8; ++r) {
            int m = m0 + r + h * 8;
            At[(size_t)m * NN + n] = f2bf(__expf(acc[j][r] - rm) * ri);
        }
    }
}

// ---------- K6: inv_colsum[m] = 1/(1e-9 + sum_n attnT[m][n]) ----------
__global__ void __launch_bounds__(256)
k_colsum(const unsigned short* attnT, float* inv_colsum) {
    int b = blockIdx.y;
    int wave = (blockIdx.x * blockDim.x + threadIdx.x) >> 5;
    int lane = threadIdx.x & 31;
    int m = wave;
    const unsigned short* row = attnT + (size_t)b * NN * NN + (size_t)m * NN;
    float s = 0.0f;
    for (int i = 0; i < NN / 256; ++i) {
        uint4 q = *(const uint4*)(row + (i * 32 + lane) * 8);
        unsigned int vv[4] = {q.x, q.y, q.z, q.w};
#pragma unroll
        for (int t = 0; t < 4; ++t) {
            s += __uint_as_float((vv[t] & 0xFFFFu) << 16);
            s += __uint_as_float(vv[t] & 0xFFFF0000u);
        }
    }
#pragma unroll
    for (int off = 16; off >= 1; off >>= 1) s += __shfl_xor(s, off, 32);
    if (lane == 0) inv_colsum[(size_t)b * NN + m] = 1.0f / (1e-9f + s);
}

// ---------- K7: x_rT strip (K=2048) + delta: dT = xT - x_r * inv_colsum ----------
__global__ void __launch_bounds__(256)
k_xr_d(const unsigned short* attnT_, const unsigned short* v_,
       const float* inv_colsum, const float* xT, unsigned short* dT) {
    int b = blockIdx.y;
    int wave = (blockIdx.x * blockDim.x + threadIdx.x) >> 5;
    int lane = threadIdx.x & 31, l = lane & 15, h = lane >> 4;
    int m0 = (wave / 4) * 16;
    int c0 = (wave % 4) * 64;
    const __bf16* At = (const __bf16*)attnT_ + (size_t)b * NN * NN;
    const __bf16* V = (const __bf16*)v_ + (size_t)b * CC * NN;
    v8f acc[4] = {};
    gemm_strip16x64<NN>(At, NN, V, NN, m0, c0, l, h, acc);
    const float* ic = inv_colsum + (size_t)b * NN;
    const float* X = xT + (size_t)b * NN * CC;
    unsigned short* D = dT + (size_t)b * NN * CC;
#pragma unroll
    for (int r = 0; r < 8; ++r) {
        int m = m0 + r + h * 8;
        float sc = ic[m];
#pragma unroll
        for (int j = 0; j < 4; ++j) {
            int c = c0 + j * 16 + l;
            float xr = acc[j][r] * sc;
            D[(size_t)m * CC + c] = f2bf(X[(size_t)m * CC + c] - xr);
        }
    }
}

// ---------- K8: tT strip + BN + ReLU + residual -> featT (f32) ----------
__global__ void __launch_bounds__(256)
k_t_feat(const unsigned short* dT_, const unsigned short* Wt_,
         const float* b_t, const float* gamma, const float* beta,
         const float* xT, float* featT) {
    int b = blockIdx.y;
    int wave = (blockIdx.x * blockDim.x + threadIdx.x) >> 5;
    int lane = threadIdx.x & 31, l = lane & 15, h = lane >> 4;
    int n0 = (wave / 4) * 16;
    int o0 = (wave % 4) * 64;
    const __bf16* Dm = (const __bf16*)dT_ + (size_t)b * NN * CC;
    const __bf16* Wt = (const __bf16*)Wt_;
    v8f acc[4] = {};
    gemm_strip16x64<CC>(Dm, CC, Wt, CC, n0, o0, l, h, acc);
    const float inv = 0.99999500003749969f;  // 1/sqrt(1+1e-5)
    const float* X = xT + (size_t)b * NN * CC;
    float* F = featT + (size_t)b * NN * CC;
#pragma unroll
    for (int j = 0; j < 4; ++j) {
        int o = o0 + j * 16 + l;
        float g = gamma[o] * inv, bt = b_t[o], be = beta[o];
#pragma unroll
        for (int r = 0; r < 8; ++r) {
            int n = n0 + r + h * 8;
            float bn = (acc[j][r] + bt) * g + be;
            F[(size_t)n * CC + o] = X[(size_t)n * CC + o] + fmaxf(bn, 0.0f);
        }
    }
}

// ---------- K9: query = Wproj @ feat + bproj, wave per point ----------
__global__ void __launch_bounds__(256)
k_query(const float* featT, const float* Wp, const float* bp, float* query) {
    int gw = (blockIdx.x * blockDim.x + threadIdx.x) >> 5;
    int lane = threadIdx.x & 31;
    int b = gw / NN, n = gw % NN;
    const float* f = featT + ((size_t)b * NN + n) * CC + lane * 8;
    float4 f0 = *(const float4*)f, f1 = *(const float4*)(f + 4);
    float s[3];
#pragma unroll
    for (int r = 0; r < 3; ++r) {
        const float* w = Wp + r * CC + lane * 8;
        float4 w0 = *(const float4*)w, w1 = *(const float4*)(w + 4);
        float p = f0.x * w0.x + f0.y * w0.y + f0.z * w0.z + f0.w * w0.w +
                  f1.x * w1.x + f1.y * w1.y + f1.z * w1.z + f1.w * w1.w;
#pragma unroll
        for (int off = 16; off >= 1; off >>= 1) p += __shfl_xor(p, off, 32);
        s[r] = p;
    }
    if (lane == 0) {
#pragma unroll
        for (int r = 0; r < 3; ++r)
            query[((size_t)b * 3 + r) * NN + n] = s[r] + bp[r];
    }
}

// ---------- K10: soft projection (top-10 KNN + softmax gather) ----------
__global__ void __launch_bounds__(256)
k_softproj(const float* pc, const float* query, const float* temp, float* out) {
    extern __shared__ float smem[];
    float* px = smem;
    float* py = smem + MM;
    float* pz = smem + 2 * MM;
    float* dbuf = smem + 3 * MM;          // 8 waves * 320
    int* ibuf = (int*)(dbuf + 8 * 320);   // 8 waves * 320
    int b = blockIdx.y;
    int qbase = blockIdx.x * 64;
    int tid = threadIdx.x;
    const float* P = pc + (size_t)b * 3 * MM;
    for (int i = tid; i < MM; i += 256) {
        px[i] = P[i];
        py[i] = P[MM + i];
        pz[i] = P[2 * MM + i];
    }
    __syncthreads();
    int wave = tid >> 5, lane = tid & 31;
    float sigma = fmaxf(temp[0] * temp[0], 1e-4f) + 1e-8f;
    float invsig = 1.0f / sigma;
    float* wd = dbuf + wave * 320;
    int* wi = ibuf + wave * 320;
    for (int qq = 0; qq < 8; ++qq) {
        int n = qbase + wave * 8 + qq;
        float qx = query[((size_t)b * 3 + 0) * NN + n];
        float qy = query[((size_t)b * 3 + 1) * NN + n];
        float qz = query[((size_t)b * 3 + 2) * NN + n];
        float td[KNN];
        int ti[KNN];
#pragma unroll
        for (int j = 0; j < KNN; ++j) { td[j] = 3.0e38f; ti[j] = 0; }
        for (int i = 0; i < MM / 32; ++i) {
            int m = i * 32 + lane;
            float dx = qx - px[m], dy = qy - py[m], dz = qz - pz[m];
            float d2 = dx * dx + dy * dy + dz * dz;
#pragma unroll
            for (int j = KNN - 1; j > 0; --j) {
                bool ltp = d2 < td[j - 1];
                bool ltc = d2 < td[j];
                td[j] = ltp ? td[j - 1] : (ltc ? d2 : td[j]);
                ti[j] = ltp ? ti[j - 1] : (ltc ? m : ti[j]);
            }
            bool lt0 = d2 < td[0];
            td[0] = lt0 ? d2 : td[0];
            ti[0] = lt0 ? m : ti[0];
        }
#pragma unroll
        for (int j = 0; j < KNN; ++j) {
            wd[lane * KNN + j] = td[j];
            wi[lane * KNN + j] = ti[j];
        }
        asm volatile("s_wait_dscnt 0" ::: "memory");  // lane lists visible wave-wide
        int ptr = 0;
        float ax = 0.f, ay = 0.f, az = 0.f, wsum = 0.f, dmin = 0.f;
#pragma unroll 1
        for (int k = 0; k < KNN; ++k) {
            float myd = (ptr < KNN) ? wd[lane * KNN + ptr] : 3.0e38f;
            int myi = (ptr < KNN) ? wi[lane * KNN + ptr] : 0;
            float bd = myd;
            int bi = myi, bl = lane;
#pragma unroll
            for (int off = 16; off >= 1; off >>= 1) {
                float od = __shfl_xor(bd, off, 32);
                int oi = __shfl_xor(bi, off, 32);
                int ol = __shfl_xor(bl, off, 32);
                bool take = (od < bd) || (od == bd && ol < bl);
                bd = take ? od : bd;
                bi = take ? oi : bi;
                bl = take ? ol : bl;
            }
            if (lane == bl) ptr++;
            if (k == 0) dmin = bd;
            float w = __expf((dmin - bd) * invsig);
            wsum += w;
            ax += w * px[bi];
            ay += w * py[bi];
            az += w * pz[bi];
        }
        if (lane == 0) {
            float iw = 1.0f / wsum;
            out[((size_t)b * 3 + 0) * NN + n] = ax * iw;
            out[((size_t)b * 3 + 1) * NN + n] = ay * iw;
            out[((size_t)b * 3 + 2) * NN + n] = az * iw;
        }
    }
}

// ---------- host ----------
extern "C" void kernel_launch(void* const* d_in, const int* in_sizes, int n_in,
                              void* d_out, int out_size, void* d_ws, size_t ws_size,
                              hipStream_t stream) {
    (void)in_sizes; (void)n_in; (void)out_size; (void)ws_size;
    const float* x = (const float*)d_in[0];
    const float* pc = (const float*)d_in[1];
    const float* Wqk = (const float*)d_in[2];
    const float* Wv = (const float*)d_in[3];
    const float* bv = (const float*)d_in[4];
    const float* Wt = (const float*)d_in[5];
    const float* bt = (const float*)d_in[6];
    const float* gamma = (const float*)d_in[7];
    const float* beta = (const float*)d_in[8];
    const float* Wp = (const float*)d_in[9];
    const float* bp = (const float*)d_in[10];
    const float* temp = (const float*)d_in[11];
    float* out = (float*)d_out;

    char* w = (char*)d_ws;
    size_t off = 0;
    auto alloc = [&](size_t bytes) -> void* {
        void* p = w + off;
        off = (off + bytes + 255) & ~(size_t)255;
        return p;
    };
    float* xT = (float*)alloc((size_t)BB * NN * CC * 4);
    unsigned short* xTb = (unsigned short*)alloc((size_t)BB * NN * CC * 2);
    unsigned short* Wqkb = (unsigned short*)alloc((size_t)CC * CC * 2);
    unsigned short* Wvb = (unsigned short*)alloc((size_t)CC * CC * 2);
    unsigned short* Wtb = (unsigned short*)alloc((size_t)CC * CC * 2);
    unsigned short* yTb = (unsigned short*)alloc((size_t)BB * NN * CC * 2);
    unsigned short* vb = (unsigned short*)alloc((size_t)BB * CC * NN * 2);
    float* rowmax = (float*)alloc((size_t)BB * NN * 4);
    float* rowsum = (float*)alloc((size_t)BB * NN * 4);
    unsigned short* attnT = (unsigned short*)alloc((size_t)BB * NN * NN * 2);
    float* invcs = (float*)alloc((size_t)BB * NN * 4);
    unsigned short* dT = (unsigned short*)alloc((size_t)BB * NN * CC * 2);
    float* featT = (float*)alloc((size_t)BB * NN * CC * 4);
    float* query = (float*)alloc((size_t)BB * 3 * NN * 4);

    // K1: transpose + bf16 convert of x
    k_transpose_x<<<dim3(NN / 32, CC / 32, BB), dim3(32, 8), 0, stream>>>(x, xT, xTb);
    // K2: weights to bf16
    k_convert_w<<<dim3(CC * CC / 256), 256, 0, stream>>>(Wqk, Wv, Wt, Wqkb, Wvb, Wtb);
    // K3a: yT[n][o] = sum_c Wqk[o][c] x[c][n]   (A = xT, B = Wqk)
    k_gemm_k256<<<dim3(64, BB), 256, 0, stream>>>(
        xTb, (size_t)NN * CC, CC, Wqkb, 0, CC, nullptr,
        yTb, (size_t)NN * CC, CC, CC / 64);
    // K3b: v[o][n] = sum_c Wv[o][c] x[c][n] + bv[o]   (A = Wv, B = xT)
    k_gemm_k256<<<dim3(64, BB), 256, 0, stream>>>(
        Wvb, 0, CC, xTb, (size_t)NN * CC, CC, bv,
        vb, (size_t)CC * NN, NN, NN / 64);
    // K4: softmax row stats over energy rows
    k_row_stats<<<dim3(16, BB), 256, 0, stream>>>(yTb, rowmax, rowsum);
    // K5: attnT (bf16)
    k_attn<<<dim3(512, BB), 256, 0, stream>>>(yTb, rowmax, rowsum, attnT);
    // K6: column-sum renorm factors
    k_colsum<<<dim3(256, BB), 256, 0, stream>>>(attnT, invcs);
    // K7: x_r + delta
    k_xr_d<<<dim3(64, BB), 256, 0, stream>>>(attnT, vb, invcs, xT, dT);
    // K8: t GEMM + BN + ReLU + residual
    k_t_feat<<<dim3(64, BB), 256, 0, stream>>>(dT, Wtb, bt, gamma, beta, xT, featT);
    // K9: 3-channel query projection
    k_query<<<dim3(BB * NN / 8), 256, 0, stream>>>(featT, Wp, bp, query);
    // K10: soft projection (KNN + softmax gather)
    size_t shmem = (size_t)(3 * MM) * 4 + (size_t)8 * 320 * 4 + (size_t)8 * 320 * 4;
    k_softproj<<<dim3(NN / 64, BB), 256, shmem, stream>>>(pc, query, temp, out);
}
